// TokenGenerator_15676630630947
// MI455X (gfx1250) — compile-verified, matrix operand order checked by reading, not run on hardware
//
#include <hip/hip_runtime.h>
#include <hip/hip_bf16.h>
#include <math.h>

// ---------------------------------------------------------------------------
// Problem dims (fixed by reference)
// ---------------------------------------------------------------------------
#define S_DIM 256
#define B_DIM 8
#define T_DIM 1024
#define D_DIM 512
#define TS_DIM 512
#define V_DIM 32000

// ---------------------------------------------------------------------------
// WMMA types (CDNA5 wave32, 16x16x32 bf16 -> f32)
// ---------------------------------------------------------------------------
typedef __attribute__((ext_vector_type(16))) __bf16 v16bf;
typedef __attribute__((ext_vector_type(8)))  float  v8f;

union FragAB { v16bf v; unsigned int u[8]; };
union FragC  { v8f   v; float        f[8]; };

// GEMM macro-tile: 128x128 per workgroup, 16 waves (512 threads),
// each wave owns a 32x32 micro-tile = 2x2 WMMA tiles (4 wmma / K-step).
#define MBLK 128
#define NBLK 128
#define KBLK 32
#define KPAD 40   // bf16/row: 80B rows -> 16B-aligned for B128 LDS writes, bank-spread

enum { EPI_F32 = 0, EPI_BF16 = 1, EPI_VT = 2, EPI_RESID_F32 = 3, EPI_TANH_BF16 = 4 };

// CDNA5 async global->LDS copy (ASYNCcnt-tracked); inline asm per bridge doc.
__device__ __forceinline__ void async_cp16(void* lds, const void* gaddr)
{
    const unsigned loff = (unsigned)(unsigned long long)lds;  // LDS offset = flat[31:0]
    asm volatile("global_load_async_to_lds_b128 %0, %1, off"
                 :: "v"(loff), "v"(gaddr) : "memory");
}
__device__ __forceinline__ void wait_async0()
{
    asm volatile("s_wait_asynccnt 0x0" ::: "memory");
}

// C[m,n] = scale * (sum_k A[m,k]*W[n,k] + bias[n])  (NT GEMM, bf16 in / f32 acc)
template<int EPI>
__global__ __launch_bounds__(512)
void gemm_nt(const __bf16* __restrict__ A, long aRow, long aBatch,
             const __bf16* __restrict__ W, long bRow, long bBatch,
             float*  __restrict__ C,  long cRow,  long cBatch,
             __bf16* __restrict__ Cb, long cbRow, long cbBatch,
             const float* __restrict__ bias,
             const float* __restrict__ resid,
             float scale, int K)
{
    __shared__ __bf16 sA[2][MBLK][KPAD];
    __shared__ __bf16 sB[2][NBLK][KPAD];

    const int tid  = threadIdx.x;
    const int wave = tid >> 5;
    const int lane = tid & 31;
    const int wm   = (wave >> 2) << 5;   // 0,32,64,96
    const int wn   = (wave & 3)  << 5;   // 0,32,64,96
    const long tileM = (long)blockIdx.x * MBLK;
    const long tileN = (long)blockIdx.y * NBLK;

    const __bf16* Ab = A + (long)blockIdx.z * aBatch;
    const __bf16* Wb = W + (long)blockIdx.z * bBatch;

    // cooperative async loads: 512 threads * 16B = 8KB = 128x32 bf16 per matrix
    const int ldRow = tid >> 2;
    const int ldK   = (tid & 3) << 3;
    const __bf16* gA = Ab + (tileM + ldRow) * aRow + ldK;
    const __bf16* gB = Wb + (tileN + ldRow) * bRow + ldK;

    FragC acc[2][2];
#pragma unroll
    for (int i = 0; i < 2; ++i)
#pragma unroll
        for (int j = 0; j < 2; ++j)
#pragma unroll
            for (int e = 0; e < 8; ++e) acc[i][j].f[e] = 0.0f;

    // prologue: stage first K-panel into buffer 0
    async_cp16(&sA[0][ldRow][ldK], gA);
    async_cp16(&sB[0][ldRow][ldK], gB);

    const int r    = lane & 15;
    const int half = (lane >> 4) << 3;   // K phase: lanes 16-31 start at K+8
    int p = 0;

    for (int k0 = 0; k0 < K; k0 += KBLK) {
        wait_async0();        // own panel-copies done
        __syncthreads();      // everyone's panel-copies done -> buf p valid

        if (k0 + KBLK < K) {  // issue next panel into the other buffer (no wait)
            async_cp16(&sA[p ^ 1][ldRow][ldK], gA + (k0 + KBLK));
            async_cp16(&sB[p ^ 1][ldRow][ldK], gB + (k0 + KBLK));
        }

        // fragments per documented 16-bit A layout (B mirrors A with N rows)
        FragAB fa[2], fb[2];
#pragma unroll
        for (int i = 0; i < 2; ++i) {
#pragma unroll
            for (int j = 0; j < 8; ++j) {
                const int kk = half + ((j < 4) ? (2 * j) : (8 + 2 * j));
                fa[i].u[j] = *(const unsigned int*)(&sA[p][wm + i * 16 + r][kk]);
                fb[i].u[j] = *(const unsigned int*)(&sB[p][wn + i * 16 + r][kk]);
            }
        }
#pragma unroll
        for (int i = 0; i < 2; ++i)
#pragma unroll
            for (int j = 0; j < 2; ++j)
                acc[i][j].v = __builtin_amdgcn_wmma_f32_16x16x32_bf16(
                    false, fa[i].v, false, fb[j].v, (short)0, acc[i][j].v, false, false);
        p ^= 1;
    }

    // Epilogue: C/D layout -> VGPR r8: M = r8 (+8 for lanes 16-31), N = lane&15
    const int nl = lane & 15;
    const int mh = (lane >> 4) << 3;
#pragma unroll
    for (int i = 0; i < 2; ++i) {
#pragma unroll
        for (int j = 0; j < 2; ++j) {
#pragma unroll
            for (int r8 = 0; r8 < 8; ++r8) {
                const long mg = tileM + wm + i * 16 + r8 + mh;
                const long ng = tileN + wn + j * 16 + nl;
                float x = acc[i][j].f[r8];
                if (bias) x += bias[ng];
                x *= scale;
                if (EPI == EPI_F32) {
                    C[(long)blockIdx.z * cBatch + mg * cRow + ng] = x;
                } else if (EPI == EPI_BF16) {
                    Cb[(long)blockIdx.z * cbBatch + mg * cbRow + ng] = (__bf16)x;
                } else if (EPI == EPI_VT) {
                    // v rows are (t*B + b); store transposed as vT[b][d][t]
                    const long t = mg / B_DIM, b = mg % B_DIM;
                    Cb[b * cbBatch + ng * cbRow + t] = (__bf16)x;
                } else if (EPI == EPI_RESID_F32) {
                    x += resid[mg * cRow + ng];
                    C[mg * cRow + ng] = x;
                } else if (EPI == EPI_TANH_BF16) {
                    Cb[mg * cbRow + ng] = (__bf16)tanhf(x);
                }
            }
        }
    }
}

// ---------------------------------------------------------------------------
// f32 -> bf16 conversion pass
// ---------------------------------------------------------------------------
__global__ void cvt_f32_bf16(const float* __restrict__ in, __bf16* __restrict__ out, long n)
{
    long i = ((long)blockIdx.x * blockDim.x + threadIdx.x) * 4;
    if (i + 3 < n) {
        const float4 f = *(const float4*)(in + i);
        out[i + 0] = (__bf16)f.x;
        out[i + 1] = (__bf16)f.y;
        out[i + 2] = (__bf16)f.z;
        out[i + 3] = (__bf16)f.w;
    } else {
        for (; i < n; ++i) out[i] = (__bf16)in[i];
    }
}

// ---------------------------------------------------------------------------
// attention softmax over T (in-place f32 + bf16 copy for the ctx GEMM)
// ---------------------------------------------------------------------------
__global__ __launch_bounds__(256)
void attn_softmax(float* __restrict__ scores, __bf16* __restrict__ attn_bf,
                  const unsigned char* __restrict__ mask)
{
    __shared__ float red[256];
    const int sb = blockIdx.x;          // s*B + b
    const int b  = sb & (B_DIM - 1);
    float* row = scores + (long)sb * T_DIM;
    const int tid = threadIdx.x;

    float v[4];
    float mx = -3.0e38f;
#pragma unroll
    for (int j = 0; j < 4; ++j) {
        const int t = tid + j * 256;
        float x = row[t];
        if (mask[b * T_DIM + t]) x = -1.0e9f;
        v[j] = x;
        mx = fmaxf(mx, x);
    }
    red[tid] = mx; __syncthreads();
    for (int s = 128; s > 0; s >>= 1) {
        if (tid < s) red[tid] = fmaxf(red[tid], red[tid + s]);
        __syncthreads();
    }
    mx = red[0]; __syncthreads();

    float sum = 0.f;
#pragma unroll
    for (int j = 0; j < 4; ++j) { v[j] = __expf(v[j] - mx); sum += v[j]; }
    red[tid] = sum; __syncthreads();
    for (int s = 128; s > 0; s >>= 1) {
        if (tid < s) red[tid] += red[tid + s];
        __syncthreads();
    }
    const float inv = 1.0f / red[0];
#pragma unroll
    for (int j = 0; j < 4; ++j) {
        const int t = tid + j * 256;
        const float p = v[j] * inv;
        row[t] = p;
        attn_bf[(long)sb * T_DIM + t] = (__bf16)p;
    }
}

// ---------------------------------------------------------------------------
// LayerNorm over D=512, output bf16 for the next GEMM
// ---------------------------------------------------------------------------
__global__ __launch_bounds__(256)
void ln_rows(const float* __restrict__ xres, const float* __restrict__ g,
             const float* __restrict__ bta, __bf16* __restrict__ out)
{
    __shared__ float red[256];
    const long row = blockIdx.x;
    const float* x = xres + row * D_DIM;
    const int tid = threadIdx.x;

    const float a0 = x[tid], a1 = x[tid + 256];
    red[tid] = a0 + a1; __syncthreads();
    for (int s = 128; s > 0; s >>= 1) { if (tid < s) red[tid] += red[tid + s]; __syncthreads(); }
    const float mu = red[0] * (1.0f / D_DIM); __syncthreads();

    const float d0 = a0 - mu, d1 = a1 - mu;
    red[tid] = d0 * d0 + d1 * d1; __syncthreads();
    for (int s = 128; s > 0; s >>= 1) { if (tid < s) red[tid] += red[tid + s]; __syncthreads(); }
    const float rstd = rsqrtf(red[0] * (1.0f / D_DIM) + 1e-5f);

    out[row * D_DIM + tid]       = (__bf16)(d0 * rstd * g[tid]       + bta[tid]);
    out[row * D_DIM + tid + 256] = (__bf16)(d1 * rstd * g[tid + 256] + bta[tid + 256]);
}

// ---------------------------------------------------------------------------
// gates: softmax over the 2-wide Wdiv head (one wave32 per row)
// ---------------------------------------------------------------------------
__global__ __launch_bounds__(256)
void gate_rows(const __bf16* __restrict__ tok, const float* __restrict__ Wdiv,
               const float* __restrict__ bdiv, float* __restrict__ gen_g,
               float* __restrict__ copy_g)
{
    const int wave = threadIdx.x >> 5, lane = threadIdx.x & 31;
    const long row = (long)blockIdx.x * 8 + wave;
    const __bf16* t = tok + row * TS_DIM;
    float a0 = 0.f, a1 = 0.f;
    for (int i = lane; i < TS_DIM; i += 32) {
        const float x = (float)t[i];
        a0 += x * Wdiv[i];
        a1 += x * Wdiv[TS_DIM + i];
    }
#pragma unroll
    for (int m = 16; m > 0; m >>= 1) {
        a0 += __shfl_xor(a0, m, 32);
        a1 += __shfl_xor(a1, m, 32);
    }
    if (lane == 0) {
        const float g0 = a0 + bdiv[0], g1 = a1 + bdiv[1];
        const float mx = fmaxf(g0, g1);
        const float e0 = __expf(g0 - mx), e1 = __expf(g1 - mx);
        const float inv = 1.0f / (e0 + e1);
        gen_g[row]  = e0 * inv;
        copy_g[row] = e1 * inv;
    }
}

// ---------------------------------------------------------------------------
// per-row (max, sum-exp) over V=32000 logits (row is L2-hot on second pass)
// ---------------------------------------------------------------------------
__global__ __launch_bounds__(256)
void row_stats(const float* __restrict__ logits, float* __restrict__ rmax,
               float* __restrict__ rsum)
{
    __shared__ float red[256];
    const long row = blockIdx.x;
    const float* x = logits + row * (long)V_DIM;
    const int tid = threadIdx.x;

    float mx = -3.0e38f;
    for (int i = tid; i < V_DIM; i += 256) mx = fmaxf(mx, x[i]);
    red[tid] = mx; __syncthreads();
    for (int s = 128; s > 0; s >>= 1) { if (tid < s) red[tid] = fmaxf(red[tid], red[tid + s]); __syncthreads(); }
    mx = red[0]; __syncthreads();

    float sum = 0.f;
    for (int i = tid; i < V_DIM; i += 256) sum += __expf(x[i] - mx);
    red[tid] = sum; __syncthreads();
    for (int s = 128; s > 0; s >>= 1) { if (tid < s) red[tid] += red[tid + s]; __syncthreads(); }
    if (tid == 0) { rmax[row] = mx; rsum[row] = red[0]; }
}

// probs = gen_gate * softmax(logits)   (in place)
__global__ void probs_scale(float* __restrict__ logits, const float* __restrict__ rmax,
                            const float* __restrict__ rsum, const float* __restrict__ gen_g)
{
    const long i = (long)blockIdx.x * blockDim.x + threadIdx.x;
    const long row = i / V_DIM;
    logits[i] = gen_g[row] * __expf(logits[i] - rmax[row]) / rsum[row];
}

// probs[s,b,copy_seq[t,b]] += copy_gate * attn[s,b,t]
__global__ void copy_scatter(const float* __restrict__ attn, const float* __restrict__ copy_g,
                             const int* __restrict__ copy_seq, float* __restrict__ probs)
{
    const long i  = (long)blockIdx.x * blockDim.x + threadIdx.x;  // sb*T + t
    const long sb = i / T_DIM;
    const int  t  = (int)(i % T_DIM);
    const int  b  = (int)(sb & (B_DIM - 1));
    const int  vtok = copy_seq[(long)t * B_DIM + b];
    atomicAdd(&probs[sb * (long)V_DIM + vtok], copy_g[sb] * attn[i]);
}

__global__ void log_out(const float* __restrict__ probs, float* __restrict__ out)
{
    const long i = (long)blockIdx.x * blockDim.x + threadIdx.x;
    out[i] = logf(probs[i] + 1e-12f);
}

// ---------------------------------------------------------------------------
// launcher
// ---------------------------------------------------------------------------
extern "C" void kernel_launch(void* const* d_in, const int* in_sizes, int n_in,
                              void* d_out, int out_size, void* d_ws, size_t ws_size,
                              hipStream_t stream)
{
    const float* outs  = (const float*)d_in[0];
    const float* graph = (const float*)d_in[1];
    const unsigned char* mask = (const unsigned char*)d_in[2];
    const int*   copy_seq = (const int*)d_in[3];
    const float* Wq = (const float*)d_in[4];   const float* bq = (const float*)d_in[5];
    const float* Wk = (const float*)d_in[6];   const float* bk = (const float*)d_in[7];
    const float* Wv = (const float*)d_in[8];   const float* bv = (const float*)d_in[9];
    const float* Wo = (const float*)d_in[10];  const float* bo = (const float*)d_in[11];
    const float* ln_g = (const float*)d_in[12]; const float* ln_b = (const float*)d_in[13];
    const float* Wt = (const float*)d_in[14];  const float* bt = (const float*)d_in[15];
    const float* Wgen = (const float*)d_in[16]; const float* bgen = (const float*)d_in[17];
    const float* Wdiv = (const float*)d_in[18]; const float* bdiv = (const float*)d_in[19];
    float* out = (float*)d_out;

    const long M  = (long)S_DIM * B_DIM;   // 2048
    const long MT = (long)T_DIM * B_DIM;   // 8192

    char* ws = (char*)d_ws;
    size_t off = 0;
    auto carve = [&](size_t bytes) -> char* {
        char* p = ws + off;
        off += (bytes + 255) & ~(size_t)255;
        return p;
    };

    __bf16* outs_bf  = (__bf16*)carve(M  * D_DIM * 2);
    __bf16* graph_bf = (__bf16*)carve(MT * D_DIM * 2);
    __bf16* wq_bf   = (__bf16*)carve((long)D_DIM * D_DIM * 2);
    __bf16* wk_bf   = (__bf16*)carve((long)D_DIM * D_DIM * 2);
    __bf16* wv_bf   = (__bf16*)carve((long)D_DIM * D_DIM * 2);
    __bf16* wo_bf   = (__bf16*)carve((long)D_DIM * D_DIM * 2);
    __bf16* wt_bf   = (__bf16*)carve((long)TS_DIM * D_DIM * 2);
    __bf16* wgen_bf = (__bf16*)carve((long)V_DIM * TS_DIM * 2);
    __bf16* q_bf    = (__bf16*)carve(M  * D_DIM * 2);
    __bf16* k_bf    = (__bf16*)carve(MT * D_DIM * 2);
    __bf16* vT_bf   = (__bf16*)carve((long)B_DIM * D_DIM * T_DIM * 2);
    float*  scores  = (float*) carve(M * T_DIM * 4);       // attn f32 in place
    __bf16* attn_bf = (__bf16*)carve(M * T_DIM * 2);
    __bf16* ctx_bf  = (__bf16*)carve(M * D_DIM * 2);
    float*  res     = (float*) carve(M * D_DIM * 4);
    __bf16* ln_bf   = (__bf16*)carve(M * D_DIM * 2);
    __bf16* tok_bf  = (__bf16*)carve(M * TS_DIM * 2);
    float*  gen_g   = (float*) carve(M * 4);
    float*  copy_g  = (float*) carve(M * 4);
    float*  rmax    = (float*) carve(M * 4);
    float*  rsum    = (float*) carve(M * 4);
    float*  logits  = (float*) carve(M * (long)V_DIM * 4);

    auto cvt = [&](const float* src, __bf16* dst, long n) {
        cvt_f32_bf16<<<dim3((unsigned)((n / 4 + 255) / 256)), dim3(256), 0, stream>>>(src, dst, n);
    };
    cvt(outs,  outs_bf,  M  * D_DIM);
    cvt(graph, graph_bf, MT * D_DIM);
    cvt(Wq, wq_bf, (long)D_DIM * D_DIM);
    cvt(Wk, wk_bf, (long)D_DIM * D_DIM);
    cvt(Wv, wv_bf, (long)D_DIM * D_DIM);
    cvt(Wo, wo_bf, (long)D_DIM * D_DIM);
    cvt(Wt, wt_bf, (long)TS_DIM * D_DIM);
    cvt(Wgen, wgen_bf, (long)V_DIM * TS_DIM);

    const float qscale = 1.0f / sqrtf((float)D_DIM);
    const dim3 blk(512);

    // q = (outs @ Wq.T + bq) * scale   -> bf16
    gemm_nt<EPI_BF16><<<dim3(M / MBLK, D_DIM / NBLK, 1), blk, 0, stream>>>(
        outs_bf, D_DIM, 0, wq_bf, D_DIM, 0,
        nullptr, 0, 0, q_bf, D_DIM, 0, bq, nullptr, qscale, D_DIM);
    // k = graph @ Wk.T + bk -> bf16
    gemm_nt<EPI_BF16><<<dim3(MT / MBLK, D_DIM / NBLK, 1), blk, 0, stream>>>(
        graph_bf, D_DIM, 0, wk_bf, D_DIM, 0,
        nullptr, 0, 0, k_bf, D_DIM, 0, bk, nullptr, 1.0f, D_DIM);
    // v = graph @ Wv.T + bv -> bf16 transposed vT[b][d][t]
    gemm_nt<EPI_VT><<<dim3(MT / MBLK, D_DIM / NBLK, 1), blk, 0, stream>>>(
        graph_bf, D_DIM, 0, wv_bf, D_DIM, 0,
        nullptr, 0, 0, vT_bf, T_DIM, (long)D_DIM * T_DIM, bv, nullptr, 1.0f, D_DIM);
    // scores[s,b,t] = q_b @ k_b.T   (per-batch NT GEMM)
    gemm_nt<EPI_F32><<<dim3(S_DIM / MBLK, T_DIM / NBLK, B_DIM), blk, 0, stream>>>(
        q_bf, (long)B_DIM * D_DIM, D_DIM, k_bf, (long)B_DIM * D_DIM, D_DIM,
        scores, (long)B_DIM * T_DIM, T_DIM, nullptr, 0, 0, nullptr, nullptr, 1.0f, D_DIM);
    // softmax over T (mask applied here)
    attn_softmax<<<dim3((unsigned)M), dim3(256), 0, stream>>>(scores, attn_bf, mask);
    // ctx = attn_b @ v_b  (NT via vT)
    gemm_nt<EPI_BF16><<<dim3(S_DIM / MBLK, D_DIM / NBLK, B_DIM), blk, 0, stream>>>(
        attn_bf, (long)B_DIM * T_DIM, T_DIM, vT_bf, T_DIM, (long)D_DIM * T_DIM,
        nullptr, 0, 0, ctx_bf, (long)B_DIM * D_DIM, D_DIM, nullptr, nullptr, 1.0f, T_DIM);
    // res = outs + ctx @ Wo.T + bo   -> f32
    gemm_nt<EPI_RESID_F32><<<dim3(M / MBLK, D_DIM / NBLK, 1), blk, 0, stream>>>(
        ctx_bf, D_DIM, 0, wo_bf, D_DIM, 0,
        res, D_DIM, 0, nullptr, 0, 0, bo, outs, 1.0f, D_DIM);
    // layernorm -> bf16
    ln_rows<<<dim3((unsigned)M), dim3(256), 0, stream>>>(res, ln_g, ln_b, ln_bf);
    // tok = tanh(ln @ Wt.T + bt) -> bf16
    gemm_nt<EPI_TANH_BF16><<<dim3(M / MBLK, TS_DIM / NBLK, 1), blk, 0, stream>>>(
        ln_bf, D_DIM, 0, wt_bf, D_DIM, 0,
        nullptr, 0, 0, tok_bf, TS_DIM, 0, bt, nullptr, 1.0f, D_DIM);
    // gen/copy gates
    gate_rows<<<dim3((unsigned)(M / 8)), dim3(256), 0, stream>>>(tok_bf, Wdiv, bdiv, gen_g, copy_g);
    // logits = tok @ Wgen.T + bgen  -> f32 (dominant GEMM, Wgen bf16 L2-resident)
    gemm_nt<EPI_F32><<<dim3(M / MBLK, V_DIM / NBLK, 1), blk, 0, stream>>>(
        tok_bf, TS_DIM, 0, wgen_bf, TS_DIM, 0,
        logits, V_DIM, 0, nullptr, 0, 0, bgen, nullptr, 1.0f, TS_DIM);
    // vocab softmax stats, scale by gen gate, scatter copy probs, log
    row_stats<<<dim3((unsigned)M), dim3(256), 0, stream>>>(logits, rmax, rsum);
    probs_scale<<<dim3((unsigned)(M * (long)V_DIM / 256)), dim3(256), 0, stream>>>(
        logits, rmax, rsum, gen_g);
    copy_scatter<<<dim3((unsigned)(M * (long)T_DIM / 256)), dim3(256), 0, stream>>>(
        scores, copy_g, copy_seq, logits);
    log_out<<<dim3((unsigned)(M * (long)V_DIM / 256)), dim3(256), 0, stream>>>(logits, out);

    (void)in_sizes; (void)n_in; (void)out_size; (void)ws_size;
}
